// CutLMHeadCE_34634616275193
// MI455X (gfx1250) — matrix operand order repaired and checked by reading, not compiled, not run on hardware
//
#include <hip/hip_runtime.h>
#include <cstdint>

#define N_TOK 4096
#define HDIM  2048
#define VOCAB 128000
#define IGNORE_INDEX (-100)

#define BM 128                       // rows per workgroup
#define BN 128                       // cols per v-iteration (8 waves x 16)
#define KP 32                        // K panel (one bf16 WMMA K-slice)
#define NCHUNKS 40
#define CHUNKV (VOCAB / NCHUNKS)     // 3200
#define VITERS (CHUNKV / BN)         // 25
#define KPANELS (HDIM / KP)          // 64
#define NPAIRS ((size_t)N_TOK * (size_t)HDIM / 2)

typedef __attribute__((ext_vector_type(16))) __bf16 v16bf;
typedef __attribute__((ext_vector_type(8)))  float  v8f;
typedef __attribute__((ext_vector_type(8)))  unsigned int v8u;

static __device__ __forceinline__ unsigned f2u(float f) { return __builtin_bit_cast(unsigned, f); }
static __device__ __forceinline__ float u2f(unsigned u) { return __builtin_bit_cast(float, u); }

// fp32 pair -> packed bf16 hi pair + packed bf16 lo pair (truncate split; lo captures residual)
static __device__ __forceinline__ void split_pair(float f0, float f1, unsigned &hi, unsigned &lo) {
    unsigned u0 = f2u(f0), u1 = f2u(f1);
    unsigned h0 = u0 & 0xFFFF0000u, h1 = u1 & 0xFFFF0000u;
    hi = h1 | (h0 >> 16);
    float l0 = f0 - u2f(h0), l1 = f1 - u2f(h1);
    lo = (f2u(l1) & 0xFFFF0000u) | (f2u(l0) >> 16);
}

// A-matrix 16x32 bf16 fragment: lane half 0 -> K{0..7,16..23}, half 1 -> K{8..15,24..31}
static __device__ __forceinline__ v16bf load_fragA(const unsigned* base, int rc, int half) {
    const uint4* p = (const uint4*)(base + rc * 16 + half * 4);
    uint4 a = p[0];   // K kbase..kbase+7
    uint4 b = p[2];   // K kbase+16..kbase+23
    v8u u = {a.x, a.y, a.z, a.w, b.x, b.y, b.z, b.w};
    return __builtin_bit_cast(v16bf, u);
}

// B-matrix 32x16 bf16 fragment: lane half 0 -> K{0..7,16..23}, half 1 -> K{8..15,24..31}
static __device__ __forceinline__ v16bf load_fragB(const unsigned* base, int rc, int half) {
    const uint4* p = (const uint4*)(base + rc * 16 + half * 4);
    uint4 a = p[0];   // K base..base+7
    uint4 b = p[2];   // K base+16..base+23
    v8u u = {a.x, a.y, a.z, a.w, b.x, b.y, b.z, b.w};
    return __builtin_bit_cast(v16bf, u);
}

static __device__ __forceinline__ v8f wmma_bf16(v16bf a, v16bf b, v8f c) {
    return __builtin_amdgcn_wmma_f32_16x16x32_bf16(false, a, false, b, (short)0, c, false, false);
}

// ---------------- kernel 0: split x into bf16 hi/lo planes (done once, 32MB ws) ----------------
__global__ void __launch_bounds__(256) split_x_kernel(const float* __restrict__ x,
                                                      unsigned* __restrict__ xhi,
                                                      unsigned* __restrict__ xlo) {
    size_t i = (size_t)blockIdx.x * 256 + threadIdx.x;   // pair index
    if (i >= NPAIRS) return;
    float2 f = ((const float2*)x)[i];
    unsigned h, l;
    split_pair(f.x, f.y, h, l);
    xhi[i] = h;
    xlo[i] = l;
}

// stage one K-panel: A via async global->LDS DMA (pre-split bf16), B via fp32 load + split
static __device__ __forceinline__ void stage_panel(int t, int rb, int v0, int kp,
                                                   const float* __restrict__ W,
                                                   const unsigned* __restrict__ xhi,
                                                   const unsigned* __restrict__ xlo,
                                                   unsigned* dAhi, unsigned* dAlo,
                                                   unsigned* dBhi, unsigned* dBlo) {
#pragma unroll
    for (int i = 0; i < 2; ++i) {
        int ch  = t * 2 + i;                 // 512 x 16B chunks for hi (and lo)
        int row = ch >> 2;
        int uo  = (ch & 3) * 4;
        size_t g = (size_t)(rb * BM + row) * (HDIM / 2) + (size_t)kp * 16 + uo;
        unsigned dsh = (unsigned)(uintptr_t)&dAhi[row * 16 + uo];
        const unsigned* ph = xhi + g;
        asm volatile("global_load_async_to_lds_b128 %0, %1, off" :: "v"(dsh), "v"(ph) : "memory");
        unsigned dsl = (unsigned)(uintptr_t)&dAlo[row * 16 + uo];
        const unsigned* pl = xlo + g;
        asm volatile("global_load_async_to_lds_b128 %0, %1, off" :: "v"(dsl), "v"(pl) : "memory");
    }
    {
        int colB = t >> 1;
        int k0   = (t & 1) * 16;
        const float* wp = W + (size_t)(v0 + colB) * HDIM + (size_t)kp * KP + k0;
        if (kp + 1 < KPANELS) __builtin_prefetch(wp + KP, 0, 1);
        float4 f0 = ((const float4*)wp)[0];
        float4 f1 = ((const float4*)wp)[1];
        float4 f2 = ((const float4*)wp)[2];
        float4 f3 = ((const float4*)wp)[3];
        unsigned h[8], l[8];
        split_pair(f0.x, f0.y, h[0], l[0]);
        split_pair(f0.z, f0.w, h[1], l[1]);
        split_pair(f1.x, f1.y, h[2], l[2]);
        split_pair(f1.z, f1.w, h[3], l[3]);
        split_pair(f2.x, f2.y, h[4], l[4]);
        split_pair(f2.z, f2.w, h[5], l[5]);
        split_pair(f3.x, f3.y, h[6], l[6]);
        split_pair(f3.z, f3.w, h[7], l[7]);
        uint4* dh = (uint4*)&dBhi[colB * 16 + k0 / 2];
        dh[0] = make_uint4(h[0], h[1], h[2], h[3]);
        dh[1] = make_uint4(h[4], h[5], h[6], h[7]);
        uint4* dl = (uint4*)&dBlo[colB * 16 + k0 / 2];
        dl[0] = make_uint4(l[0], l[1], l[2], l[3]);
        dl[1] = make_uint4(l[4], l[5], l[6], l[7]);
    }
}

// ---------------- kernel 1: fused GEMM (bf16x3 WMMA) + online per-chunk logsumexp ----------------
__global__ void __launch_bounds__(256) lmhead_partial_kernel(const float* __restrict__ W,
                                                             const unsigned* __restrict__ xhi,
                                                             const unsigned* __restrict__ xlo,
                                                             const int* __restrict__ tgt,
                                                             float* __restrict__ tgt_logit,
                                                             float2* __restrict__ partials) {
    __shared__ unsigned sAhi[2][BM * 16];   // 2 x 8KB (double-buffered K panels)
    __shared__ unsigned sAlo[2][BM * 16];
    __shared__ unsigned sBhi[2][BN * 16];
    __shared__ unsigned sBlo[2][BN * 16];
    __shared__ float    sMax[8][BM];        // per-wave running row max
    __shared__ float    sSum[8][BM];        // per-wave running row sum-exp
    __shared__ int      sTgt[BM];

    const int rb    = blockIdx.x;           // row block (fast-varying -> chunk-mates share L2)
    const int chunk = blockIdx.y;
    const int t     = threadIdx.x;
    const int wv    = t >> 5;
    const int lane  = t & 31;
    const int half  = lane >> 4;
    const int idx   = lane & 15;
    const float NEG_INF = -__builtin_inff();

    for (int r = t; r < BM; r += 256) sTgt[r] = tgt[rb * BM + r];
    for (int i = t; i < 8 * BM; i += 256) { (&sMax[0][0])[i] = NEG_INF; (&sSum[0][0])[i] = 0.f; }
    __syncthreads();

    for (int vi = 0; vi < VITERS; ++vi) {
        const int v0 = chunk * CHUNKV + vi * BN;

        v8f c[8];
        const v8f vzero = {0.f, 0.f, 0.f, 0.f, 0.f, 0.f, 0.f, 0.f};
#pragma unroll
        for (int i = 0; i < 8; ++i) c[i] = vzero;

        // prologue: stage panel 0 into buffer 0
        stage_panel(t, rb, v0, 0, W, xhi, xlo, sAhi[0], sAlo[0], sBhi[0], sBlo[0]);
        asm volatile("s_wait_asynccnt 0" ::: "memory");
        __syncthreads();

        for (int kp = 0; kp < KPANELS; ++kp) {
            const int cur = kp & 1;
            const int nxt = cur ^ 1;

            // overlap: stage panel kp+1 into the other buffer while computing panel kp
            if (kp + 1 < KPANELS)
                stage_panel(t, rb, v0, kp + 1, W, xhi, xlo, sAhi[nxt], sAlo[nxt], sBhi[nxt], sBlo[nxt]);

            v16bf bh = load_fragB(sBhi[cur], wv * 16 + idx, half);
            v16bf bl = load_fragB(sBlo[cur], wv * 16 + idx, half);
#pragma unroll
            for (int tt = 0; tt < 8; ++tt) {
                v16bf ah = load_fragA(sAhi[cur], tt * 16 + idx, half);
                v16bf al = load_fragA(sAlo[cur], tt * 16 + idx, half);
                c[tt] = wmma_bf16(ah, bh, c[tt]);   // hi*hi
                c[tt] = wmma_bf16(ah, bl, c[tt]);   // hi*lo
                c[tt] = wmma_bf16(al, bh, c[tt]);   // lo*hi
            }

            // fence next panel's async A DMA; barrier also orders B ds_stores and buffer reuse
            asm volatile("s_wait_asynccnt 0" ::: "memory");
            __syncthreads();
        }

        // ---- epilogue: online logsumexp over this 128-col tile + target-logit capture ----
        const int col = v0 + wv * 16 + idx;
#pragma unroll
        for (int tt = 0; tt < 8; ++tt) {
            const int rbase = tt * 16 + half * 8;
            float m[8], s[8];
#pragma unroll
            for (int r = 0; r < 8; ++r) m[r] = c[tt][r];
#pragma unroll
            for (int mask = 1; mask < 16; mask <<= 1)
#pragma unroll
                for (int r = 0; r < 8; ++r) m[r] = fmaxf(m[r], __shfl_xor(m[r], mask, 16));
#pragma unroll
            for (int r = 0; r < 8; ++r) s[r] = __expf(c[tt][r] - m[r]);
#pragma unroll
            for (int mask = 1; mask < 16; mask <<= 1)
#pragma unroll
                for (int r = 0; r < 8; ++r) s[r] += __shfl_xor(s[r], mask, 16);
#pragma unroll
            for (int r = 0; r < 8; ++r)
                if (col == sTgt[rbase + r]) tgt_logit[rb * BM + rbase + r] = c[tt][r];
            if (idx == 0) {
#pragma unroll
                for (int r = 0; r < 8; ++r) {
                    float mo = sMax[wv][rbase + r], so = sSum[wv][rbase + r];
                    float mn = fmaxf(mo, m[r]);
                    sSum[wv][rbase + r] = so * __expf(mo - mn) + s[r] * __expf(m[r] - mn);
                    sMax[wv][rbase + r] = mn;
                }
            }
        }
    }

    __syncthreads();
    // merge the 8 per-wave states and emit this chunk's (max, sumexp) partial per row
    for (int r = t; r < BM; r += 256) {
        float m = sMax[0][r], s = sSum[0][r];
#pragma unroll
        for (int w = 1; w < 8; ++w) {
            float pm = sMax[w][r], ps = sSum[w][r];
            float mn = fmaxf(m, pm);
            s = s * __expf(m - mn) + ps * __expf(pm - mn);
            m = mn;
        }
        partials[(size_t)chunk * N_TOK + rb * BM + r] = make_float2(m, s);
    }
}

// ---------------- kernel 2: merge chunks, CE, masked mean ----------------
__global__ void __launch_bounds__(256) ce_finalize_kernel(const float2* __restrict__ partials,
                                                          const float* __restrict__ tgt_logit,
                                                          const int* __restrict__ tgt,
                                                          float* __restrict__ out) {
    __shared__ float sAcc[256];
    __shared__ int   sCnt[256];
    const int t = threadIdx.x;
    float acc = 0.f;
    int   cnt = 0;
    for (int row = t; row < N_TOK; row += 256) {
        float m = -__builtin_inff(), s = 0.f;
        for (int ck = 0; ck < NCHUNKS; ++ck) {
            float2 p = partials[(size_t)ck * N_TOK + row];
            float mn = fmaxf(m, p.x);
            s = s * __expf(m - mn) + p.y * __expf(p.x - mn);
            m = mn;
        }
        float lse = m + __logf(s);
        int tg = tgt[row];
        if (tg != IGNORE_INDEX) { acc += lse - tgt_logit[row]; cnt += 1; }
    }
    sAcc[t] = acc; sCnt[t] = cnt;
    __syncthreads();
    for (int off = 128; off > 0; off >>= 1) {
        if (t < off) { sAcc[t] += sAcc[t + off]; sCnt[t] += sCnt[t + off]; }
        __syncthreads();
    }
    if (t == 0) out[0] = sAcc[0] / (float)(sCnt[0] > 0 ? sCnt[0] : 1);
}

extern "C" void kernel_launch(void* const* d_in, const int* in_sizes, int n_in,
                              void* d_out, int out_size, void* d_ws, size_t ws_size,
                              hipStream_t stream) {
    (void)in_sizes; (void)n_in; (void)out_size; (void)ws_size;
    const float* x   = (const float*)d_in[0];
    const float* W   = (const float*)d_in[1];
    const int*   tgt = (const int*)d_in[2];
    float* out = (float*)d_out;

    unsigned* xhi = (unsigned*)d_ws;                 // NPAIRS uints (16MB)
    unsigned* xlo = xhi + NPAIRS;                    // NPAIRS uints (16MB)
    float* tlg    = (float*)(xlo + NPAIRS);          // N_TOK floats
    float2* parts = (float2*)(tlg + N_TOK);          // NCHUNKS * N_TOK float2

    split_x_kernel<<<(unsigned)(NPAIRS / 256), 256, 0, stream>>>(x, xhi, xlo);
    lmhead_partial_kernel<<<dim3(N_TOK / BM, NCHUNKS), 256, 0, stream>>>(W, xhi, xlo, tgt, tlg, parts);
    ce_finalize_kernel<<<1, 256, 0, stream>>>(parts, tlg, tgt, out);
}
